// MuSHIN_82351702933507
// MI455X (gfx1250) — matrix operand order repaired and checked by reading, not compiled, not run on hardware
//
#include <hip/hip_runtime.h>

typedef __attribute__((ext_vector_type(16))) __bf16 v16bf;
typedef __attribute__((ext_vector_type(8)))  float  v8f;

#define NNODES 10000
#define MEDGES 5000
#define FIN    512
#define EMB    256
#define CONVD  128
#define NHEADS 3
#define DEG    32
#define HC     (NHEADS * CONVD)     /* 384 */
#define NENTRY (MEDGES * DEG)       /* 160000 */
#define NEG_SLOPE 0.2f

// ---------------------------------------------------------------------------
// WMMA GEMM: C[M x NDIM] = act(A[M x K] @ B[K x NDIM] + bias)
// fp32 in/out, bf16 compute, fp32 accumulate (v_wmma_f32_16x16x32_bf16).
// 256 threads = 8 waves. Block tile 128(M) x 64(N); each wave owns 16x64
// (4 accumulators -> 4 WMMA per k-step, A fragment reused 4x).
// B slab (32K x 64N) is staged once per block into LDS, pre-converted to
// bf16 and pre-swizzled into the exact WMMA B-fragment register layout, so
// readers do pure 32B LDS fragment loads. NDIM is a template param so all
// global strides fold into immediate offsets (no 64-bit VALU address chains).
// ---------------------------------------------------------------------------
template <int NDIM, bool HAS_BIAS, bool RELU>
__global__ void gemm_bf16_wmma(const float* __restrict__ A,
                               const float* __restrict__ B,
                               const float* __restrict__ bias,
                               float* __restrict__ C,
                               int Mdim, int Kdim)
{
    // [n-frag][lane][k-element] ; one row = one lane's v16bf fragment (32B)
    __shared__ __align__(32) __bf16 Bt[4][32][16];

    const int tid  = threadIdx.x;
    const int lane = tid & 31;
    const int wid  = tid >> 5;            // 0..7 -> M sub-tile
    const int r    = lane & 15;
    const int half = lane >> 4;

    const int m0 = blockIdx.y * 128 + wid * 16;
    const int n0 = blockIdx.x * 64;

    int mrow = m0 + r;
    if (mrow > Mdim - 1) mrow = Mdim - 1;     // clamp ragged M tiles (stores guarded)
    const float* Arow = A + (size_t)mrow * Kdim;

    // B staging work split: thread t handles rows kk & kk+16, 4 consecutive cols
    const int kk    = tid >> 4;               // 0..15
    const int nf    = (tid & 15) * 4;         // 0,4,...,60
    const int wfrag = nf >> 4;                // all 4 cols land in one 16-col frag
    const int wr    = nf & 15;

    v8f acc[4] = {{}, {}, {}, {}};

    for (int k0 = 0; k0 < Kdim; k0 += 32) {
        // ---- stage B slab: coalesced float4 global reads, bf16 convert ----
        const float4 b_lo = *(const float4*)(B + (size_t)(k0 + kk)      * NDIM + n0 + nf);
        const float4 b_hi = *(const float4*)(B + (size_t)(k0 + kk + 16) * NDIM + n0 + nf);
        if (k0 + 32 < Kdim)   // gfx1250 global_prefetch_b8 for the next slab
            __builtin_prefetch(B + (size_t)(k0 + 32 + kk) * NDIM + n0 + nf, 0, 1);

        __syncthreads();      // previous iteration's fragment reads complete
        const float* plo = (const float*)&b_lo;
        const float* phi = (const float*)&b_hi;
        #pragma unroll
        for (int j = 0; j < 4; ++j) {
            // element (k,c): lane = (k>=16)*16 + (c&15), elem = k&15
            Bt[wfrag][wr + j     ][kk] = (__bf16)plo[j];   // k = kk      (half 0)
            Bt[wfrag][wr + j + 16][kk] = (__bf16)phi[j];   // k = kk + 16 (half 1)
        }
        __syncthreads();

        // ---- A fragment straight from global (merges into b128 loads) ----
        // layout: lanes 0-15: K 0..15 ; lanes 16-31: K 16..31
        v16bf a;
        const float* Ap = Arow + k0 + half * 16;
        #pragma unroll
        for (int i = 0; i < 16; ++i) {
            const int kidx = i;
            a[i] = (__bf16)Ap[kidx];
        }

        // ---- 4 ready-made B fragments from LDS (2x ds_load_b128 each) ----
        v16bf b0 = *(const v16bf*)Bt[0][lane];
        v16bf b1 = *(const v16bf*)Bt[1][lane];
        v16bf b2 = *(const v16bf*)Bt[2][lane];
        v16bf b3 = *(const v16bf*)Bt[3][lane];

        acc[0] = __builtin_amdgcn_wmma_f32_16x16x32_bf16(false, a, false, b0,
                                                         (short)0, acc[0], false, false);
        acc[1] = __builtin_amdgcn_wmma_f32_16x16x32_bf16(false, a, false, b1,
                                                         (short)0, acc[1], false, false);
        acc[2] = __builtin_amdgcn_wmma_f32_16x16x32_bf16(false, a, false, b2,
                                                         (short)0, acc[2], false, false);
        acc[3] = __builtin_amdgcn_wmma_f32_16x16x32_bf16(false, a, false, b3,
                                                         (short)0, acc[3], false, false);
    }

    // ---- epilogue: bias / relu, guarded ragged-M stores ----
    #pragma unroll
    for (int f = 0; f < 4; ++f) {
        const int gcol = n0 + f * 16 + r;
        float bv = 0.0f;
        if (HAS_BIAS) bv = bias[gcol];
        #pragma unroll
        for (int rr = 0; rr < 8; ++rr) {
            const int grow = m0 + rr + half * 8;   // C/D: VGPR rr -> M = rr + half*8
            if (grow < Mdim) {
                float v = acc[f][rr];
                if (HAS_BIAS) v += bv;
                if (RELU) v = fmaxf(v, 0.0f);
                C[(size_t)grow * NDIM + gcol] = v;
            }
        }
    }
}

// he_attr[e, :] = b_attr + sum_k W_attr[node(e,k), :]     (inc_T @ W_attr, sparse)
// Async-DMA version: the 32 scattered 1KB rows are copied global->LDS with
// GLOBAL_LOAD_ASYNC_TO_LDS_B128 (ASYNCcnt machinery, no VGPR round-trip),
// each wave drains its own ASYNCcnt with s_wait_asynccnt, then the workgroup
// reduces out of LDS. 32KB of the 320KB/WGP LDS pool.
__global__ void edge_gather_attr(const int* __restrict__ nodes,
                                 const float* __restrict__ Wattr,
                                 const float* __restrict__ battr,
                                 float* __restrict__ he)
{
    __shared__ int nid[DEG];
    __shared__ __align__(16) float rows[DEG][EMB];   // 32 KB staging
    const int e = blockIdx.x, t = threadIdx.x;       // 256 threads = 8 waves
    const int lane = t & 31, w = t >> 5;
    if (t < DEG) nid[t] = nodes[e * DEG + t];
    __syncthreads();

    // wave-relative LDS byte address of the staging buffer (addrspace(3) cast)
    const unsigned ldsbase = (unsigned)(unsigned long long)
        (__attribute__((address_space(3))) float*)&rows[0][0];

    // wave w copies rows 4w..4w+3; per row: 32 lanes x 16B x 2 passes = 1KB.
    // INST_OFFSET applies to BOTH the global and the LDS address (ISA 08 §4.4).
    #pragma unroll
    for (int q = 0; q < 4; ++q) {
        const int row = w * 4 + q;
        const unsigned long long src =
            (unsigned long long)(const void*)(Wattr + (size_t)nid[row] * EMB + lane * 4);
        const unsigned dst = ldsbase + (unsigned)(row * (EMB * 4) + lane * 16);
        asm volatile("global_load_async_to_lds_b128 %0, %1, off"
                     :: "v"(dst), "v"(src) : "memory");
        asm volatile("global_load_async_to_lds_b128 %0, %1, off offset:512"
                     :: "v"(dst), "v"(src) : "memory");
    }
    asm volatile("s_wait_asynccnt 0x0" ::: "memory");  // drain this wave's copies
    __syncthreads();                                   // all waves' rows visible

    float s = battr[t];
    #pragma unroll 4
    for (int k = 0; k < DEG; ++k) s += rows[k][t];
    he[(size_t)e * EMB + t] = s;
}

// s[row, h] = dot(v[row, h, :], att[h, attOff : attOff+128])
__global__ void attn_score(const float* __restrict__ v,
                           const float* __restrict__ att, int attOff,
                           float* __restrict__ s, int rows)
{
    int idx = blockIdx.x * blockDim.x + threadIdx.x;
    if (idx >= rows * NHEADS) return;
    int row = idx / NHEADS, h = idx % NHEADS;
    const float* vp = v + (size_t)row * HC + h * CONVD;
    const float* ap = att + h * (2 * CONVD) + attOff;
    float acc = 0.f;
    #pragma unroll 8
    for (int c = 0; c < CONVD; ++c) acc += vp[c] * ap[c];
    s[idx] = acc;
}

__global__ void zero_f32(float* __restrict__ p, int n)
{
    int i = blockIdx.x * blockDim.x + threadIdx.x;
    if (i < n) p[i] = 0.0f;
}

// One wave per edge: leaky-relu + per-edge softmax over its DEG=32 entries
// (wave32 == DEG: pure shuffle reduction). Also counts node degree Dn.
__global__ void alpha_softmax(const int* __restrict__ nodes,
                              const float* __restrict__ sx,
                              const float* __restrict__ se,
                              float* __restrict__ alpha,
                              float* __restrict__ Dn)
{
    const int e = blockIdx.x;
    const int k = threadIdx.x;                       // 0..31
    const int entry = e * DEG + k;
    const int node = nodes[entry];
    #pragma unroll
    for (int h = 0; h < NHEADS; ++h) {
        float a = sx[node * NHEADS + h] + se[e * NHEADS + h];
        a = (a > 0.f) ? a : NEG_SLOPE * a;
        float m = a;
        #pragma unroll
        for (int off = 16; off > 0; off >>= 1) m = fmaxf(m, __shfl_xor(m, off, 32));
        float ex = __expf(a - m);
        float sum = ex;
        #pragma unroll
        for (int off = 16; off > 0; off >>= 1) sum += __shfl_xor(sum, off, 32);
        alpha[entry * NHEADS + h] = ex / sum;
    }
    atomicAdd(&Dn[node], 1.0f);
}

// edge_out[e, h, c] = (1/32) * sum_k alpha[e,k,h] * x[node(e,k), h, c]
__global__ void edge_aggregate(const int* __restrict__ nodes,
                               const float* __restrict__ alpha,
                               const float* __restrict__ x,
                               float* __restrict__ eo)
{
    __shared__ int nid[DEG];
    __shared__ float al[DEG * NHEADS];
    const int e = blockIdx.x, t = threadIdx.x;       // 384 threads
    if (t < DEG) nid[t] = nodes[e * DEG + t];
    if (t >= 128 && t < 128 + DEG * NHEADS) al[t - 128] = alpha[e * DEG * NHEADS + (t - 128)];
    __syncthreads();
    const int h = t >> 7;                            // t / CONVD
    float s = 0.f;
    #pragma unroll 4
    for (int k = 0; k < DEG; ++k)
        s += al[k * NHEADS + h] * x[(size_t)nid[k] * HC + t];
    eo[(size_t)e * HC + t] = s * (1.0f / (float)DEG);
}

// node_out[n, h, c] += (1/Dn[n]) * alpha[entry,h] * edge_out[e, h, c]
__global__ void node_scatter(const int* __restrict__ nodes,
                             const float* __restrict__ alpha,
                             const float* __restrict__ eo,
                             const float* __restrict__ Dn,
                             float* __restrict__ no)
{
    const int entry = blockIdx.x;                    // 160000 blocks
    const int t = threadIdx.x;                       // 384 threads
    const int e = entry >> 5;
    const int node = nodes[entry];
    const int h = t >> 7;
    const float w = alpha[entry * NHEADS + h] / Dn[node];
    atomicAdd(&no[(size_t)node * HC + t], w * eo[(size_t)e * HC + t]);
}

// he_feat[e, :] = sum_k (node_out[node(e,k), :] + b_conv)  ==  sum + 32*b_conv
__global__ void edge_gather_out(const int* __restrict__ nodes,
                                const float* __restrict__ no,
                                const float* __restrict__ bconv,
                                float* __restrict__ hf)
{
    __shared__ int nid[DEG];
    const int e = blockIdx.x, t = threadIdx.x;       // 384 threads
    if (t < DEG) nid[t] = nodes[e * DEG + t];
    __syncthreads();
    float s = (float)DEG * bconv[t];
    #pragma unroll 4
    for (int k = 0; k < DEG; ++k) s += no[(size_t)nid[k] * HC + t];
    hf[(size_t)e * HC + t] = s;
}

// out[e, 0:2] = he_feat[e, :] @ W_out + b_out
__global__ void final_proj(const float* __restrict__ hf,
                           const float* __restrict__ Wout,
                           const float* __restrict__ bout,
                           float* __restrict__ out)
{
    const int e = blockIdx.x * blockDim.x + threadIdx.x;
    if (e >= MEDGES) return;
    float o0 = bout[0], o1 = bout[1];
    const float* p = hf + (size_t)e * HC;
    #pragma unroll 8
    for (int c = 0; c < HC; ++c) {
        float f = p[c];
        o0 += f * Wout[2 * c];
        o1 += f * Wout[2 * c + 1];
    }
    out[2 * e]     = o0;
    out[2 * e + 1] = o1;
}

extern "C" void kernel_launch(void* const* d_in, const int* in_sizes, int n_in,
                              void* d_out, int out_size, void* d_ws, size_t ws_size,
                              hipStream_t stream)
{
    const float* in_feat = (const float*)d_in[0];
    /* d_in[1] = dense incidence matrix: unused (edge_index carries structure) */
    const int*   nodes   = (const int*)d_in[2];      // edge_index row 0 = node ids
    const float* W_enc   = (const float*)d_in[3];
    const float* b_enc   = (const float*)d_in[4];
    const float* W_attr  = (const float*)d_in[5];
    const float* b_attr  = (const float*)d_in[6];
    const float* W_lin   = (const float*)d_in[7];
    const float* att     = (const float*)d_in[8];
    const float* b_conv  = (const float*)d_in[9];
    const float* W_out   = (const float*)d_in[10];
    const float* b_out   = (const float*)d_in[11];
    float* out = (float*)d_out;

    // workspace carve-out (floats), ~71 MB total
    float* w   = (float*)d_ws;
    float* x0  = w;                            // N x EMB
    float* xf  = x0  + (size_t)NNODES * EMB;   // N x 384
    float* hea = xf  + (size_t)NNODES * HC;    // M x EMB
    float* eaf = hea + (size_t)MEDGES * EMB;   // M x 384
    float* sx  = eaf + (size_t)MEDGES * HC;    // N x 3
    float* se  = sx  + (size_t)NNODES * NHEADS;// M x 3
    float* alp = se  + (size_t)MEDGES * NHEADS;// E x 3
    float* Dn  = alp + (size_t)NENTRY * NHEADS;// N
    float* eo  = Dn  + (size_t)NNODES;         // M x 384
    float* no  = eo  + (size_t)MEDGES * HC;    // N x 384
    float* hf  = no  + (size_t)NNODES * HC;    // M x 384

    // 1) x0 = relu(in_feat @ W_enc + b_enc)            [10000x512x256, WMMA]
    gemm_bf16_wmma<EMB, true, true>
        <<<dim3(EMB / 64, (NNODES + 127) / 128), 256, 0, stream>>>(
            in_feat, W_enc, b_enc, x0, NNODES, FIN);

    // 2) he_attr = inc_T @ W_attr + b_attr             [async-LDS 32-gather]
    edge_gather_attr<<<MEDGES, EMB, 0, stream>>>(nodes, W_attr, b_attr, hea);

    // 3) x = x0 @ W_lin                                 [10000x256x384, WMMA]
    gemm_bf16_wmma<HC, false, false>
        <<<dim3(HC / 64, (NNODES + 127) / 128), 256, 0, stream>>>(
            x0, W_lin, nullptr, xf, NNODES, EMB);

    // 4) ea = he_attr @ W_lin                           [5000x256x384, WMMA]
    gemm_bf16_wmma<HC, false, false>
        <<<dim3(HC / 64, (MEDGES + 127) / 128), 256, 0, stream>>>(
            hea, W_lin, nullptr, eaf, MEDGES, EMB);

    // 5) factored attention scores: sx[n,h]=x.att_x, se[e,h]=ea.att_e
    attn_score<<<(NNODES * NHEADS + 255) / 256, 256, 0, stream>>>(xf, att, 0, sx, NNODES);
    attn_score<<<(MEDGES * NHEADS + 255) / 256, 256, 0, stream>>>(eaf, att, CONVD, se, MEDGES);

    // 6) node degree counts + per-edge wave32 softmax
    zero_f32<<<(NNODES + 255) / 256, 256, 0, stream>>>(Dn, NNODES);
    alpha_softmax<<<MEDGES, DEG, 0, stream>>>(nodes, sx, se, alp, Dn);

    // 7) edge_out = segment_sum(Be * alpha * x_i)
    edge_aggregate<<<MEDGES, HC, 0, stream>>>(nodes, alp, xf, eo);

    // 8) node_out = segment_sum(Dn^-1 * alpha * edge_out[edge]) via atomics
    zero_f32<<<((NNODES * HC) + 255) / 256, 256, 0, stream>>>(no, NNODES * HC);
    node_scatter<<<NENTRY, HC, 0, stream>>>(nodes, alp, eo, Dn, no);

    // 9) he_feat = inc_T @ (node_out + b_conv)          [sparse 32-gather]
    edge_gather_out<<<MEDGES, HC, 0, stream>>>(nodes, no, b_conv, hf);

    // 10) out = he_feat @ W_out + b_out                 [N=2, scalar]
    final_proj<<<(MEDGES + 255) / 256, 256, 0, stream>>>(hf, W_out, b_out, out);

    (void)in_sizes; (void)n_in; (void)out_size; (void)ws_size;
}
